// Attention_85761906966847
// MI455X (gfx1250) — matrix operand order repaired and checked by reading, not compiled, not run on hardware
//
#include <hip/hip_runtime.h>
#include <math.h>

#define BATCH 8
#define SLEN  2048
#define DDIM  512
#define LDQ   520                          // padded LDS row stride (elems) for 16x512 tiles
#define SCORE_SCALE 0.04419417382415922f   // 1/sqrt(512)

typedef __attribute__((ext_vector_type(16))) __bf16 bf16x16;
typedef __attribute__((ext_vector_type(8)))  __bf16 bf16x8;
typedef __attribute__((ext_vector_type(8)))  float  f32x8;
typedef __attribute__((ext_vector_type(4)))  unsigned int u32x4;
typedef __attribute__((ext_vector_type(8)))  int  i32x8;
typedef __attribute__((ext_vector_type(4)))  int  i32x4;

#if __has_builtin(__builtin_amdgcn_tensor_load_to_lds) && __has_builtin(__builtin_amdgcn_s_wait_tensorcnt)
#define HAVE_TDM 1
#else
#define HAVE_TDM 0
#endif

__device__ __forceinline__ bf16x16 join8(bf16x8 lo, bf16x8 hi) {
    return __builtin_shufflevector(lo, hi, 0,1,2,3,4,5,6,7,8,9,10,11,12,13,14,15);
}

// A fragment (16x32 bf16) from the padded LDS tile (row stride LDQ elems).
// Lanes 0-15: row m, K = 0..7 then 16..23 ; lanes 16-31: K = 8..15 then 24..31.
__device__ __forceinline__ bf16x16 load_a_lds(const __bf16* base, int k, int lane) {
    int m  = lane & 15;
    int hi = (lane >> 4) << 3;
    const __bf16* p = base + m * LDQ + k + hi;
    return join8(*(const bf16x8*)p, *(const bf16x8*)(p + 16));
}

// B fragment (32x16 bf16): element (k,n) at base + n*ld + k (columns contiguous).
// Lane n (0-15) holds K=0..7 then 16..23 of column n ; lane 16+n holds K=8..15 then 24..31.
__device__ __forceinline__ bf16x16 load_b_colcontig(const __bf16* base, int ld, int lane) {
    int n  = lane & 15;
    int ko = (lane >> 4) << 3;
    const __bf16* p = base + (size_t)n * ld + ko;
    return join8(*(const bf16x8*)p, *(const bf16x8*)(p + 16));
}

__device__ __forceinline__ f32x8 wmma_bf16(bf16x16 a, bf16x16 b, f32x8 c) {
    return __builtin_amdgcn_wmma_f32_16x16x32_bf16(false, a, false, b, (short)0, c,
                                                   false, false);
}

// Load a 16x512 bf16 tile (memory row stride = DDIM elems) into LDS with a
// 16B pad per row (LDS row stride = LDQ elems). TDM when available.
__device__ __forceinline__ void tile_load_16x512(const __bf16* gsrc, __bf16* ldst,
                                                 int tidx) {
#if HAVE_TDM
    if (tidx < 32) {                      // one wave issues the DMA
        unsigned long long ga = (unsigned long long)(uintptr_t)gsrc;
        unsigned ldsoff = (unsigned)(uintptr_t)ldst;   // LDS addr = low 32 bits
        u32x4 g0;
        g0[0] = 1u;                                        // count=1 user desc
        g0[1] = ldsoff;                                    // lds_addr
        g0[2] = (unsigned)(ga & 0xffffffffu);              // global_addr lo
        g0[3] = (unsigned)((ga >> 32) & 0x01ffffffu) | (2u << 30);  // hi | type=2
        i32x8 g1;
        g1[0] = (1 << 16) | (1 << 20) | (7 << 22) | (3 << 25); // 2B elems, pad: 4DW per 256DW
        g1[1] = (int)(512u << 16);        // tensor_dim0 = 512
        g1[2] = (int)(16u  << 16);        // tensor_dim1 = 16
        g1[3] = (int)(512u << 16);        // tile_dim0 = 512
        g1[4] = 16;                       // tile_dim1 = 16, tile_dim2 = 0
        g1[5] = 512;                      // tensor_dim0_stride = DDIM
        g1[6] = 0;
        g1[7] = 0;
        i32x4 z4 = {0, 0, 0, 0};
        i32x8 z8 = {0, 0, 0, 0, 0, 0, 0, 0};
        __builtin_amdgcn_tensor_load_to_lds(g0, g1, z4, z4, z8, 0);
        __builtin_amdgcn_s_wait_tensorcnt((short)0);
    }
#else
    for (int c = tidx; c < 1024; c += 128) {      // 1024 x 16B chunks
        int row = c >> 6;
        int off = (c & 63) << 3;
        *(bf16x8*)&ldst[row * LDQ + off] = *(const bf16x8*)&gsrc[row * DDIM + off];
    }
#endif
    __syncthreads();
}

// ---------------------------------------------------------------- convert ---
__global__ void cvt_f32_bf16_kernel(const float* __restrict__ src,
                                    __bf16* __restrict__ dst, int n) {
    int i = blockIdx.x * blockDim.x + threadIdx.x;
    int stride = gridDim.x * blockDim.x;
    for (; i < n; i += stride) dst[i] = (__bf16)src[i];
}

// --------------------------------------------------------------- projection -
// out = X @ W^T + bias. 4 waves/block, wave owns 64 output cols; 2 blocks/row tile.
__global__ __launch_bounds__(128) void proj_kernel(
        const __bf16* __restrict__ X,
        const __bf16* __restrict__ W,
        const float*  __restrict__ bias,
        __bf16* __restrict__ outRow,   // [B*S, D] (Q/K) when transposed==0
        __bf16* __restrict__ outT,     // [B, D, S] (V)  when transposed==1
        int transposed) {
    __shared__ __align__(32) __bf16 s_x[16][LDQ];
    int tidx = threadIdx.x;
    int lane = tidx & 31, wave = tidx >> 5;
    int row0 = (blockIdx.x >> 1) * 16;
    int col0 = (blockIdx.x & 1) * 256 + wave * 64;

    tile_load_16x512(X + (size_t)row0 * DDIM, &s_x[0][0], tidx);
    const __bf16* sx = &s_x[0][0];

    f32x8 zero = {0.f,0.f,0.f,0.f,0.f,0.f,0.f,0.f};
    f32x8 acc[4];
#pragma unroll
    for (int t = 0; t < 4; ++t) acc[t] = zero;

    // software-pipelined K loop: fetch chunk k+1 while WMMAs consume chunk k
    bf16x16 a_cur = load_a_lds(sx, 0, lane);
    bf16x16 b_cur[4];
#pragma unroll
    for (int t = 0; t < 4; ++t)
        b_cur[t] = load_b_colcontig(W + (size_t)(col0 + t * 16) * DDIM, DDIM, lane);

#pragma unroll
    for (int kc = 0; kc < 16; ++kc) {
        bf16x16 a_nxt = a_cur;
        bf16x16 b_nxt[4];
        if (kc < 15) {
            int k = (kc + 1) * 32;
            a_nxt = load_a_lds(sx, k, lane);
#pragma unroll
            for (int t = 0; t < 4; ++t)
                b_nxt[t] = load_b_colcontig(W + (size_t)(col0 + t * 16) * DDIM + k,
                                            DDIM, lane);
        } else {
#pragma unroll
            for (int t = 0; t < 4; ++t) b_nxt[t] = b_cur[t];
        }
#pragma unroll
        for (int t = 0; t < 4; ++t) acc[t] = wmma_bf16(a_cur, b_cur[t], acc[t]);
        a_cur = a_nxt;
#pragma unroll
        for (int t = 0; t < 4; ++t) b_cur[t] = b_nxt[t];
    }

    int n15  = lane & 15;
    int moff = (lane >> 4) << 3;
#pragma unroll
    for (int t = 0; t < 4; ++t) {
        int   col = col0 + t * 16 + n15;
        float bn  = bias[col];
#pragma unroll
        for (int r = 0; r < 8; ++r) {
            float v   = acc[t][r] + bn;
            int   row = row0 + r + moff;
            if (!transposed) {
                outRow[(size_t)row * DDIM + col] = (__bf16)v;
            } else {
                int bb = row / SLEN, s = row % SLEN;
                outT[((size_t)bb * DDIM + col) * SLEN + s] = (__bf16)v;
            }
        }
    }
}

// ------------------------------------------------------------------- flash --
// One block (4 waves) per 16-query tile; online softmax over 64-key blocks.
// Wave w: score keys [k0+16w, k0+16w+16) ; output dims [128w, 128w+128).
__global__ __launch_bounds__(128) void flash_kernel(
        const __bf16* __restrict__ Qb,   // [B*S, D]
        const __bf16* __restrict__ Kb,   // [B*S, D]
        const __bf16* __restrict__ Vt,   // [B, D, S]
        const int*    __restrict__ lengths,
        float* __restrict__ out) {       // [B*S, D]
    __shared__ __align__(32) __bf16 s_q[16][LDQ];      // staged Q tile (padded)
    __shared__ __align__(32) float  s_scores[16][65];  // +1 pad: conflict-free col reads
    __shared__ __align__(32) __bf16 s_p[16][72];       // 36 DW/row: distinct bank starts
    __shared__ float s_m[16], s_l[16], s_scale[16];

    int tidx = threadIdx.x;
    int lane = tidx & 31;
    int wave = tidx >> 5;
    int qt   = blockIdx.x % (SLEN / 16);
    int b    = blockIdx.x / (SLEN / 16);
    int q0   = qt * 16;
    int len  = lengths[b];

    const __bf16* Kbase = Kb + (size_t)b * SLEN * DDIM;
    const __bf16* Vbase = Vt + (size_t)b * DDIM * SLEN;

    if (tidx < 16) { s_m[tidx] = -__builtin_inff(); s_l[tidx] = 0.f; }
    tile_load_16x512(Qb + ((size_t)b * SLEN + q0) * DDIM, &s_q[0][0], tidx);
    const __bf16* sq = &s_q[0][0];

    int klast = (q0 + 15 < len - 1) ? (q0 + 15) : (len - 1);   // len >= 1
    int nblk  = klast / 64 + 1;

    f32x8 zero = {0.f,0.f,0.f,0.f,0.f,0.f,0.f,0.f};
    f32x8 acc[8];
#pragma unroll
    for (int t = 0; t < 8; ++t) acc[t] = zero;

    int dbase = wave * 128;
    int n15   = lane & 15;
    int moff  = (lane >> 4) << 3;

    for (int blk = 0; blk < nblk; ++blk) {
        int k0 = blk * 64;
        int kw = k0 + wave * 16;

        if (blk + 1 < nblk)
            __builtin_prefetch(Kbase + (size_t)(kw + 64) * DDIM, 0, 1);

        // ---- scores: S = Q(16xD) @ K_sub^T -> 16x16 fp32 (pipelined) -------
        f32x8 sacc = zero;
        bf16x16 a_cur = load_a_lds(sq, 0, lane);
        bf16x16 b_cur = load_b_colcontig(Kbase + (size_t)kw * DDIM, DDIM, lane);
#pragma unroll
        for (int kc = 0; kc < 16; ++kc) {
            bf16x16 a_nxt = a_cur, b_nxt = b_cur;
            if (kc < 15) {
                int k = (kc + 1) * 32;
                a_nxt = load_a_lds(sq, k, lane);
                b_nxt = load_b_colcontig(Kbase + (size_t)kw * DDIM + k, DDIM, lane);
            }
            sacc = wmma_bf16(a_cur, b_cur, sacc);
            a_cur = a_nxt; b_cur = b_nxt;
        }

        // mask + stash (D layout: VGPR r -> row r+moff, lane -> col n15)
#pragma unroll
        for (int r = 0; r < 8; ++r) {
            int  m   = r + moff;
            int  key = kw + n15;
            bool ok  = (key <= q0 + m) && (key < len);
            s_scores[m][wave * 16 + n15] = ok ? sacc[r] * SCORE_SCALE
                                             : -__builtin_inff();
        }
        __syncthreads();

        // ---- online softmax stats (wave 0) --------------------------------
        if (wave == 0) {
            int   row = n15, h = lane >> 4;
            float mx = -__builtin_inff();
            for (int j = 0; j < 32; ++j) mx = fmaxf(mx, s_scores[row][h * 32 + j]);
            mx = fmaxf(mx, __shfl_xor(mx, 16));
            float m_old = s_m[row];
            float m_new = fmaxf(m_old, mx);
            float sum = 0.f;
            for (int j = 0; j < 32; ++j) {
                float e = __expf(s_scores[row][h * 32 + j] - m_new);
                sum += e;
                s_p[row][h * 32 + j] = (__bf16)e;
            }
            sum += __shfl_xor(sum, 16);
            if (h == 0) {
                float sc = __expf(m_old - m_new);
                s_scale[row] = sc;
                s_m[row]     = m_new;
                s_l[row]     = s_l[row] * sc + sum;
            }
        }
        __syncthreads();

        // ---- rescale accumulators, then O += P @ V ------------------------
        float sc[8];
#pragma unroll
        for (int r = 0; r < 8; ++r) sc[r] = s_scale[r + moff];
#pragma unroll
        for (int t = 0; t < 8; ++t)
#pragma unroll
            for (int r = 0; r < 8; ++r) acc[t][r] *= sc[r];

#pragma unroll
        for (int c = 0; c < 2; ++c) {
            int mrow = lane & 15;
            int hi   = (lane >> 4) << 3;
            const __bf16* pp = &s_p[mrow][c * 32 + hi];
            bf16x16 a = join8(*(const bf16x8*)pp, *(const bf16x8*)(pp + 16));
            bf16x16 bf[8];
#pragma unroll
            for (int t = 0; t < 8; ++t)
                bf[t] = load_b_colcontig(
                    Vbase + (size_t)(dbase + t * 16) * SLEN + k0 + c * 32,
                    SLEN, lane);
#pragma unroll
            for (int t = 0; t < 8; ++t) acc[t] = wmma_bf16(a, bf[t], acc[t]);
        }
        __syncthreads();
    }

    // ---- epilogue: O /= l, store fp32 -------------------------------------
    float linv[8];
#pragma unroll
    for (int r = 0; r < 8; ++r) linv[r] = 1.0f / s_l[r + moff];
    float* obase = out + ((size_t)b * SLEN + q0) * DDIM;
#pragma unroll
    for (int t = 0; t < 8; ++t)
#pragma unroll
        for (int r = 0; r < 8; ++r)
            obase[(size_t)(r + moff) * DDIM + dbase + t * 16 + n15] =
                acc[t][r] * linv[r];
}

// ------------------------------------------------------------------ launch --
extern "C" void kernel_launch(void* const* d_in, const int* in_sizes, int n_in,
                              void* d_out, int out_size, void* d_ws, size_t ws_size,
                              hipStream_t stream) {
    const float* x   = (const float*)d_in[0];
    const float* wq  = (const float*)d_in[1];
    const float* bq  = (const float*)d_in[2];
    const float* wk  = (const float*)d_in[3];
    const float* bk  = (const float*)d_in[4];
    const float* wv  = (const float*)d_in[5];
    const float* bv  = (const float*)d_in[6];
    const int*   len = (const int*)d_in[7];
    float* out = (float*)d_out;

    char* ws = (char*)d_ws;
    const size_t XB = (size_t)BATCH * SLEN * DDIM * 2;   // 16 MB bf16
    const size_t WB = (size_t)DDIM * DDIM * 2;           // 512 KB bf16
    __bf16* Xb  = (__bf16*)(ws);
    __bf16* Wqb = (__bf16*)(ws + XB);
    __bf16* Wkb = (__bf16*)(ws + XB + WB);
    __bf16* Wvb = (__bf16*)(ws + XB + 2 * WB);
    __bf16* Qb  = (__bf16*)(ws + XB + 3 * WB);
    __bf16* Kb  = (__bf16*)(ws + 2 * XB + 3 * WB);
    __bf16* Vt  = (__bf16*)(ws + 3 * XB + 3 * WB);

    int nX = BATCH * SLEN * DDIM;
    int nW = DDIM * DDIM;
    cvt_f32_bf16_kernel<<<dim3(4096), 256, 0, stream>>>(x,  Xb,  nX);
    cvt_f32_bf16_kernel<<<dim3(256),  256, 0, stream>>>(wq, Wqb, nW);
    cvt_f32_bf16_kernel<<<dim3(256),  256, 0, stream>>>(wk, Wkb, nW);
    cvt_f32_bf16_kernel<<<dim3(256),  256, 0, stream>>>(wv, Wvb, nW);

    dim3 pgrid(BATCH * SLEN / 16 * 2);
    proj_kernel<<<pgrid, 128, 0, stream>>>(Xb, Wqb, bq, Qb, nullptr, 0);
    proj_kernel<<<pgrid, 128, 0, stream>>>(Xb, Wkb, bk, Kb, nullptr, 0);
    proj_kernel<<<pgrid, 128, 0, stream>>>(Xb, Wvb, bv, nullptr, Vt, 1);

    flash_kernel<<<dim3(BATCH * (SLEN / 16)), 128, 0, stream>>>(Qb, Kb, Vt, len, out);
}
